// LocalPoolPointnet_28913719837031
// MI455X (gfx1250) — matrix operand order repaired and checked by reading, not compiled, not run on hardware
//
#include <hip/hip_runtime.h>
#include <hip/hip_bf16.h>
#include <stdint.h>

// ---------------- problem constants (match reference) ----------------
#define BB   4
#define NN   32768
#define MM   (BB * NN)        // 131072 rows
#define HH   128
#define K2H  256
#define RR   32
#define SS   (RR * RR * RR)   // 32768
#define CC   128
#define NBLK 5
#define MT   4                // 16-row M-tiles per workgroup (64 rows)
#define ROWS (MT * 16)

// ---------------- vector / fragment types ----------------
typedef __attribute__((ext_vector_type(4)))  unsigned int  u32x4;
typedef __attribute__((ext_vector_type(8)))  float         v8f;
typedef __attribute__((ext_vector_type(16))) __bf16        v16bf;

union Frag {
    v16bf bf;
    u32x4 q[2];
};

// ---------------- small helpers ----------------
__device__ __forceinline__ unsigned short f2bf(float f) {
    unsigned int u = __float_as_uint(f);
    u += 0x7fffu + ((u >> 16) & 1u);          // round-to-nearest-even
    return (unsigned short)(u >> 16);
}
__device__ __forceinline__ float bf2f(unsigned short h) {
    return __uint_as_float(((unsigned int)h) << 16);
}
__device__ __forceinline__ void atomicMaxF(float* a, float v) {
    if (v >= 0.0f) atomicMax((int*)a, __float_as_int(v));
    else           atomicMin((unsigned int*)a, __float_as_uint(v));
}

// A-fragment (16x32 bf16) from row-major LDS tile.
// Layout per ISA 7.12.2: lanes 0-15 -> K 0..7 & 16..23; lanes 16-31 -> K 8..15 & 24..31.
__device__ __forceinline__ Frag load_af(const unsigned short* tile, int stride,
                                        int kt, int lane) {
    const int m  = lane & 15;
    const int kb = (lane < 16) ? 0 : 8;
    const unsigned short* p = tile + m * stride + kt * 32 + kb;
    Frag f;
    f.q[0] = *(const u32x4*)(p);          // K = kb .. kb+7
    f.q[1] = *(const u32x4*)(p + 16);     // K = kb+16 .. kb+23
    return f;
}

// relu on packed bf16 pairs via native VOP3P packed max: max(x, +0.0)
__device__ __forceinline__ v16bf relu_bf(Frag f) {
#pragma unroll
    for (int j = 0; j < 2; ++j) {
#pragma unroll
        for (int k = 0; k < 4; ++k) {
            unsigned int r;
            asm("v_pk_max_num_bf16 %0, %1, 0" : "=v"(r) : "v"(f.q[j][k]));
            f.q[j][k] = r;
        }
    }
    return f.bf;
}

// B-fragment (32x16 bf16) from pre-packed weights:
// pack[((k/32)*128 + n)*32 + (k%32)] -> per-lane k%32 = kb..kb+7 and kb+16..kb+23 (kb = 0 or 8).
__device__ __forceinline__ v16bf load_b(const unsigned short* pack, int kt,
                                        int nl, int lane) {
    const int kb = (lane < 16) ? 0 : 8;
    const unsigned short* p = pack + ((size_t)(kt * 128 + nl)) * 32 + kb;
    Frag f;
    f.q[0] = *(const u32x4*)(p);
    f.q[1] = *(const u32x4*)(p + 16);
    return f.bf;
}

// ---------------- kernels ----------------
__global__ void fill_kernel(float* __restrict__ p, float v, size_t n) {
    size_t i = (size_t)blockIdx.x * 256 + threadIdx.x;
    if (i < n) p[i] = v;
}

// f32 (K x 128) row-major -> bf16 B-fragment pack
__global__ void pack_kernel(const float* __restrict__ src,
                            unsigned short* __restrict__ dst, int K) {
    int e = blockIdx.x * 256 + threadIdx.x;
    if (e >= K * 128) return;
    int k = e >> 7, n = e & 127;
    dst[((size_t)((k >> 5) * 128 + n)) * 32 + (k & 31)] = f2bf(src[e]);
}

// per-point: coords, voxel index, residual, gidx/count/mask, fc_pos -> X (bf16 M x 256)
__global__ void prep_kernel(const float* __restrict__ p,
                            const float* __restrict__ fpw,   // (3,256)
                            const float* __restrict__ fpb,   // (256)
                            unsigned short* __restrict__ X,
                            int* __restrict__ index,
                            int* __restrict__ gidx,
                            float* __restrict__ cnt,
                            float* __restrict__ maskOut) {
    int bn = blockIdx.x * 256 + threadIdx.x;
    if (bn >= MM) return;
    int b = bn >> 15;  // / NN
    const float inv = 1.0f / 1.101f;   // 1 + PAD + 1e-3
    float cx = fminf(fmaxf(p[bn * 3 + 0] * inv + 0.5f, 0.0f), 0.999f);
    float cy = fminf(fmaxf(p[bn * 3 + 1] * inv + 0.5f, 0.0f), 0.999f);
    float cz = fminf(fmaxf(p[bn * 3 + 2] * inv + 0.5f, 0.0f), 0.999f);
    int ix = (int)(cx * RR), iy = (int)(cy * RR), iz = (int)(cz * RR);
    index[bn] = ix + RR * (iy + RR * iz);
    // residual coords (in-place semantics of reference)
    float rx = cx - (floorf(cx * RR) + 0.5f) / RR;
    float ry = cy - (floorf(cy * RR) + 0.5f) / RR;
    float rz = cz - (floorf(cz * RR) + 0.5f) / RR;
    int gx = (int)(rx * RR), gy = (int)(ry * RR), gz = (int)(rz * RR);
    int gi = gx + RR * (gy + RR * gz);
    if (gi < 0) gi = 0; if (gi >= SS) gi = SS - 1;
    gidx[bn] = gi;
    atomicAdd(&cnt[(size_t)b * SS + gi], 1.0f);
    maskOut[(((size_t)b * RR + gz) * RR + gy) * RR + gx] = 1.0f;
    // fc_pos: (rx,ry,rz) @ (3x256) + b
    unsigned short* xr = X + (size_t)bn * K2H;
    for (int j = 0; j < K2H; ++j) {
        float acc = fpb[j] + rx * fpw[j] + ry * fpw[256 + j] + rz * fpw[512 + j];
        xr[j] = f2bf(acc);
    }
}

// ResnetBlockFC: Y[:, :128] = X @ ws + relu(relu(X)@w0 + b0) @ w1 + b1  (in-place into X)
// 64 rows per WG: each wave holds 4 M-tiles, reusing each B-fragment 4x.
__global__ __launch_bounds__(256) void resblock_kernel(
        const unsigned short* __restrict__ X,    // M x 256 bf16
        const unsigned short* __restrict__ w0p,  // packed 256x128
        const float* __restrict__ b0,
        const unsigned short* __restrict__ w1p,  // packed 128x128
        const float* __restrict__ b1,
        const unsigned short* __restrict__ wsp,  // packed 256x128
        unsigned short* __restrict__ Y) {
    __shared__ unsigned short Xt[ROWS * 256];   // 32 KB
    __shared__ unsigned short Nt[ROWS * 128];   // 16 KB
    const int tid = threadIdx.x;
    const int row0 = blockIdx.x * ROWS;
    // stage 64x256 bf16 tile (2048 16B vectors, 8 per thread)
    {
        const u32x4* src = (const u32x4*)(X + (size_t)row0 * 256);
        u32x4* dst = (u32x4*)Xt;
#pragma unroll
        for (int i = 0; i < 8; ++i) dst[tid + i * 256] = src[tid + i * 256];
        // prefetch next WG's tile into L2 (global_prefetch_b8)
        if (row0 + ROWS < MM)
            __builtin_prefetch(X + (size_t)(row0 + ROWS) * 256 + tid * 64, 0, 1);
    }
    __syncthreads();
    const int wave = tid >> 5, lane = tid & 31;
    const int nl   = wave * 16 + (lane & 15);
    const int moff = (lane < 16) ? 0 : 8;
    v8f acc0[MT] = {};   // relu(x)@w0 path
    v8f accS[MT] = {};   // x@ws shortcut path
#pragma unroll
    for (int kt = 0; kt < 8; ++kt) {          // K = 256
        v16bf bf0 = load_b(w0p, kt, nl, lane);
        v16bf bfs = load_b(wsp, kt, nl, lane);
#pragma unroll
        for (int mt = 0; mt < MT; ++mt) {
            Frag fa = load_af(Xt + mt * 16 * 256, 256, kt, lane);
            v16bf aR = relu_bf(fa);
            acc0[mt] = __builtin_amdgcn_wmma_f32_16x16x32_bf16(
                false, aR, false, bf0, (short)0, acc0[mt], false, false);
            accS[mt] = __builtin_amdgcn_wmma_f32_16x16x32_bf16(
                false, fa.bf, false, bfs, (short)0, accS[mt], false, false);
        }
    }
    const float bias0 = b0[nl];
#pragma unroll
    for (int mt = 0; mt < MT; ++mt) {
#pragma unroll
        for (int v = 0; v < 8; ++v) {
            float t = acc0[mt][v] + bias0;
            t = t > 0.0f ? t : 0.0f;          // relu(net)
            Nt[(mt * 16 + v + moff) * 128 + nl] = f2bf(t);
        }
    }
    __syncthreads();
#pragma unroll
    for (int kt = 0; kt < 4; ++kt) {          // K = 128
        v16bf bf1 = load_b(w1p, kt, nl, lane);
#pragma unroll
        for (int mt = 0; mt < MT; ++mt) {
            Frag fa = load_af(Nt + mt * 16 * 128, 128, kt, lane);
            accS[mt] = __builtin_amdgcn_wmma_f32_16x16x32_bf16(
                false, fa.bf, false, bf1, (short)0, accS[mt], false, false);
        }
    }
    const float bias1 = b1[nl];
#pragma unroll
    for (int mt = 0; mt < MT; ++mt) {
#pragma unroll
        for (int v = 0; v < 8; ++v) {
            float t = accS[mt][v] + bias1;
            Y[(size_t)(row0 + mt * 16 + v + moff) * 256 + nl] = f2bf(t);
        }
    }
}

__global__ void scatter_max_kernel(const unsigned short* __restrict__ X,
                                   const int* __restrict__ index,
                                   float* __restrict__ fea) {
    size_t id = (size_t)blockIdx.x * 256 + threadIdx.x;   // M*32 threads
    int bn = (int)(id >> 5);
    int ch = ((int)id & 31) * 4;
    int b  = bn >> 15;
    int ix = index[bn];
    float* dst = fea + ((size_t)(b * SS + ix)) * 128 + ch;
    const unsigned short* src = X + (size_t)bn * 256 + ch;
#pragma unroll
    for (int j = 0; j < 4; ++j) atomicMaxF(dst + j, bf2f(src[j]));
}

__global__ void gather_concat_kernel(const float* __restrict__ fea,
                                     const int* __restrict__ index,
                                     unsigned short* __restrict__ X) {
    size_t id = (size_t)blockIdx.x * 256 + threadIdx.x;   // M*32 threads
    int bn = (int)(id >> 5);
    int ch = ((int)id & 31) * 4;
    int b  = bn >> 15;
    int ix = index[bn];
    const float* src = fea + ((size_t)(b * SS + ix)) * 128 + ch;
    unsigned short* dst = X + (size_t)bn * 256 + 128 + ch;
#pragma unroll
    for (int j = 0; j < 4; ++j) {
        float v = src[j];
        if (v == -__builtin_huge_valf()) v = 0.0f;        // empty bin -> 0
        dst[j] = f2bf(v);
    }
}

// fc_c GEMM + scatter-mean numerator straight into d_out
__global__ __launch_bounds__(256) void fcc_kernel(
        const unsigned short* __restrict__ X,   // net in cols 0..127, stride 256
        const unsigned short* __restrict__ wp,  // packed 128x128
        const float* __restrict__ bias,
        const int* __restrict__ gidx,
        float* __restrict__ outFea) {           // (B, C, S)
    __shared__ unsigned short Nt[ROWS * 128];   // 16 KB
    const int tid = threadIdx.x;
    const int row0 = blockIdx.x * ROWS;
    {
        // 64 rows x 128 cols = 1024 16B vectors, 4 per thread
#pragma unroll
        for (int i = 0; i < 4; ++i) {
            int v = tid + i * 256;
            int r = v >> 4, c8 = v & 15;
            ((u32x4*)Nt)[v] = *(const u32x4*)(X + (size_t)(row0 + r) * 256 + c8 * 8);
        }
    }
    __syncthreads();
    const int wave = tid >> 5, lane = tid & 31;
    const int nl   = wave * 16 + (lane & 15);
    const int moff = (lane < 16) ? 0 : 8;
    v8f acc[MT] = {};
#pragma unroll
    for (int kt = 0; kt < 4; ++kt) {
        v16bf b = load_b(wp, kt, nl, lane);
#pragma unroll
        for (int mt = 0; mt < MT; ++mt) {
            Frag fa = load_af(Nt + mt * 16 * 128, 128, kt, lane);
            acc[mt] = __builtin_amdgcn_wmma_f32_16x16x32_bf16(
                false, fa.bf, false, b, (short)0, acc[mt], false, false);
        }
    }
    const float bia = bias[nl];
#pragma unroll
    for (int mt = 0; mt < MT; ++mt) {
#pragma unroll
        for (int v = 0; v < 8; ++v) {
            int row = row0 + mt * 16 + v + moff;
            int b   = row >> 15;
            int g   = gidx[row];
            atomicAdd(&outFea[((size_t)(b * CC + nl)) * SS + g], acc[mt][v] + bia);
        }
    }
}

__global__ void finalize_kernel(float* __restrict__ outFea,
                                const float* __restrict__ cnt) {
    size_t e = (size_t)blockIdx.x * 256 + threadIdx.x;
    if (e >= (size_t)BB * CC * SS) return;
    int s = (int)(e % SS);
    int b = (int)(e / ((size_t)CC * SS));
    outFea[e] /= fmaxf(cnt[(size_t)b * SS + s], 1.0f);
}

// ---------------- launch ----------------
extern "C" void kernel_launch(void* const* d_in, const int* in_sizes, int n_in,
                              void* d_out, int out_size, void* d_ws, size_t ws_size,
                              hipStream_t stream) {
    (void)in_sizes; (void)n_in; (void)out_size; (void)ws_size;
    const float* p        = (const float*)d_in[0];
    const float* fc_pos_w = (const float*)d_in[1];
    const float* fc_pos_b = (const float*)d_in[2];
    const float* fc0_w    = (const float*)d_in[3];
    const float* fc0_b    = (const float*)d_in[4];
    const float* fc1_w    = (const float*)d_in[5];
    const float* fc1_b    = (const float*)d_in[6];
    const float* sc_w     = (const float*)d_in[7];
    const float* fc_c_w   = (const float*)d_in[8];
    const float* fc_c_b   = (const float*)d_in[9];

    float* outFea  = (float*)d_out;                      // B*C*S
    float* outMask = outFea + (size_t)BB * CC * SS;      // B*S (as float 0/1)

    // workspace layout
    char* w = (char*)d_ws;
    unsigned short* X = (unsigned short*)w;  w += (size_t)MM * K2H * 2;       // 64 MB
    float* fea = (float*)w;                  w += (size_t)BB * SS * HH * 4;   // 64 MB
    int* index = (int*)w;                    w += (size_t)MM * 4;
    int* gidx  = (int*)w;                    w += (size_t)MM * 4;
    float* cnt = (float*)w;                  w += (size_t)BB * SS * 4;
    unsigned short* w0p = (unsigned short*)w; w += (size_t)NBLK * K2H * HH * 2;
    unsigned short* w1p = (unsigned short*)w; w += (size_t)NBLK * HH * HH * 2;
    unsigned short* wsp = (unsigned short*)w; w += (size_t)NBLK * K2H * HH * 2;
    unsigned short* fcp = (unsigned short*)w; w += (size_t)HH * HH * 2;

    const size_t outTotal = (size_t)BB * CC * SS + (size_t)BB * SS;
    fill_kernel<<<(int)((outTotal + 255) / 256), 256, 0, stream>>>((float*)d_out, 0.0f, outTotal);
    fill_kernel<<<(BB * SS + 255) / 256, 256, 0, stream>>>(cnt, 0.0f, (size_t)BB * SS);

    // pack weights -> bf16 fragment layout
    for (int i = 0; i < NBLK; ++i) {
        pack_kernel<<<(K2H * HH + 255) / 256, 256, 0, stream>>>(fc0_w + (size_t)i * K2H * HH,
                                                                w0p + (size_t)i * K2H * HH, K2H);
        pack_kernel<<<(HH * HH + 255) / 256, 256, 0, stream>>>(fc1_w + (size_t)i * HH * HH,
                                                               w1p + (size_t)i * HH * HH, HH);
        pack_kernel<<<(K2H * HH + 255) / 256, 256, 0, stream>>>(sc_w + (size_t)i * K2H * HH,
                                                                wsp + (size_t)i * K2H * HH, K2H);
    }
    pack_kernel<<<(HH * HH + 255) / 256, 256, 0, stream>>>(fc_c_w, fcp, HH);

    prep_kernel<<<MM / 256, 256, 0, stream>>>(p, fc_pos_w, fc_pos_b, X, index, gidx, cnt, outMask);

    // block 0
    resblock_kernel<<<MM / ROWS, 256, 0, stream>>>(X, w0p, fc0_b, w1p, fc1_b, wsp, X);

    // blocks 1..4: pool -> concat -> resblock
    const int sgGrid = (int)(((size_t)MM * 32) / 256);
    for (int i = 1; i < NBLK; ++i) {
        fill_kernel<<<(int)(((size_t)BB * SS * HH + 255) / 256), 256, 0, stream>>>(
            fea, -__builtin_huge_valf(), (size_t)BB * SS * HH);
        scatter_max_kernel<<<sgGrid, 256, 0, stream>>>(X, index, fea);
        gather_concat_kernel<<<sgGrid, 256, 0, stream>>>(fea, index, X);
        resblock_kernel<<<MM / ROWS, 256, 0, stream>>>(X,
            w0p + (size_t)i * K2H * HH, fc0_b + (size_t)i * HH,
            w1p + (size_t)i * HH * HH,  fc1_b + (size_t)i * HH,
            wsp + (size_t)i * K2H * HH, X);
    }

    // fc_c + scatter-mean numerator, then divide by counts
    fcc_kernel<<<MM / ROWS, 256, 0, stream>>>(X, fcp, fc_c_b, gidx, outFea);
    finalize_kernel<<<(int)(((size_t)BB * CC * SS + 255) / 256), 256, 0, stream>>>(outFea, cnt);
}
